// PhiAsLossEngine_31825707664172
// MI455X (gfx1250) — matrix operand order repaired and checked by reading, not compiled, not run on hardware
//
#include <hip/hip_runtime.h>

// ---------------------------------------------------------------- types
typedef __bf16 bf16;
typedef bf16 v16bf __attribute__((ext_vector_type(16)));
typedef bf16 v8bf  __attribute__((ext_vector_type(8)));
typedef float v8f  __attribute__((ext_vector_type(8)));

#define N_CELLS 32768
#define IN_DIM  256
#define HID_DIM 512
#define OUT_DIM 256
#define MLP_DIM 128
#define N_F     8
#define FS      (N_CELLS / N_F)  // 4096

// ------------------------------------------------------------- helpers
__device__ __forceinline__ bf16 f2bf(float f) {
  unsigned u = __float_as_uint(f);
  unsigned r = (u + 0x7fffu + ((u >> 16) & 1u)) >> 16;
  unsigned short s = (unsigned short)r;
  return *(bf16*)&s;
}

__device__ __forceinline__ float sigmoidf_(float x) {
  return 1.0f / (1.0f + __expf(-x));
}

__device__ __forceinline__ v8f zero8() {
  v8f z = {0.f, 0.f, 0.f, 0.f, 0.f, 0.f, 0.f, 0.f};
  return z;
}

// A fragment: 16x32 (MxK) bf16, row-major, leading dim ld.
// lanes 0-15: M=lane, K chunks [0..7](v0-3) and [16..23](v4-7);
// lanes 16-31: K chunks [8..15] and [24..31].  (ISA 7.12.2)
__device__ __forceinline__ v16bf load_a_frag(const bf16* base, int ld, int row0,
                                             int k0, int lane) {
  const int m  = lane & 15;
  const int hi = lane >> 4;
  const bf16* p = base + (size_t)(row0 + m) * ld + k0 + hi * 8;
  v16bf r;
  ((v8bf*)&r)[0] = *(const v8bf*)(p);
  ((v8bf*)&r)[1] = *(const v8bf*)(p + 16);
  return r;
}

// B fragment: 32x16 (KxN); column n = row n of row-major W[N,K] (X @ W.T).
// lanes 0-15 hold K=[0..7],[16..23] of col lane&15, lanes 16-31 hold K=[8..15],[24..31].
// Works for both global and LDS base pointers (addrspace inferred on inline).
__device__ __forceinline__ v16bf load_b_frag(const bf16* base, int ld, int n0,
                                             int k0, int lane) {
  const int n  = lane & 15;
  const int hi = lane >> 4;
  const bf16* p = base + (size_t)(n0 + n) * ld + k0 + hi * 8;
  v16bf r;
  ((v8bf*)&r)[0] = *(const v8bf*)(p);
  ((v8bf*)&r)[1] = *(const v8bf*)(p + 16);
  return r;
}

#define WMMA_BF16(a, b, c) \
  __builtin_amdgcn_wmma_f32_16x16x32_bf16(false, (a), false, (b), (short)0, (c), false, false)

// ------------------------------------------------------------- convert
__global__ void k_f2bf(const float* __restrict__ src, bf16* __restrict__ dst, int n) {
  int i = blockIdx.x * blockDim.x + threadIdx.x;
  int stride = gridDim.x * blockDim.x;
  for (; i < n; i += stride) dst[i] = f2bf(src[i]);
}

__global__ void k_f2bf_sub(const float* __restrict__ src, bf16* __restrict__ dst,
                           int rows, int cols, int src_ld, int col0) {
  int n = rows * cols;
  int i = blockIdx.x * blockDim.x + threadIdx.x;
  int stride = gridDim.x * blockDim.x;
  for (; i < n; i += stride) {
    int r = i / cols, c = i - r * cols;
    dst[i] = f2bf(src[(size_t)r * src_ld + col0 + c]);
  }
}

// -------------------------------------------------- x-vector precompute
__global__ void k_xvec(const float* __restrict__ x,
                       const float* __restrict__ Wa1, const float* __restrict__ ba1,
                       const float* __restrict__ Wg1, const float* __restrict__ bg1,
                       const float* __restrict__ ba2, const float* __restrict__ bg2,
                       float* __restrict__ va, float* __restrict__ vg,
                       float* __restrict__ bdelta) {
  int t = threadIdx.x;
  if (t < MLP_DIM) {
    float sa = ba1[t], sg = bg1[t];
    const float* ra = Wa1 + (size_t)t * (IN_DIM + HID_DIM);
    const float* rg = Wg1 + (size_t)t * (IN_DIM + HID_DIM);
    for (int k = 0; k < IN_DIM; ++k) {
      float xv = x[k];
      sa = fmaf(ra[k], xv, sa);
      sg = fmaf(rg[k], xv, sg);
    }
    va[t] = sa;
    vg[t] = sg;
  }
  if (t < OUT_DIM) bdelta[t] = ba2[t] - bg2[t];
}

// ------------------------------------------------------------- MLP L1
// H1 = relu(h @ Wa1h.T + va) bf16 ; G1 = -relu(h @ Wg1h.T + vg) bf16
// grid (N/512, 2), block 256, dynamic LDS = 128KB (full W for this engine).
// Each block: stage W once, process 4 row-tiles of 128 rows.
#define MLP1_ROWREP 4
__global__ void __launch_bounds__(256) k_mlp1(
    const bf16* __restrict__ hidbf, const bf16* __restrict__ wa1h,
    const bf16* __restrict__ wg1h, const float* __restrict__ va,
    const float* __restrict__ vg, bf16* __restrict__ h1, bf16* __restrict__ g1) {
  extern __shared__ char smem_raw[];
  bf16* sW = (bf16*)smem_raw;  // [128][512]

  const int lane = threadIdx.x & 31;
  const int wave = threadIdx.x >> 5;
  const int eng  = blockIdx.y;
  const bf16*  W   = eng ? wg1h : wa1h;
  const float* vb  = eng ? vg : va;
  bf16*        dst = eng ? g1 : h1;
  const float  sgn = eng ? -1.0f : 1.0f;

  // cooperative LDS fill: 64K bf16 = 8192 uint4
  for (int i = threadIdx.x; i < (MLP_DIM * HID_DIM) / 8; i += 256)
    ((uint4*)sW)[i] = ((const uint4*)W)[i];
  __syncthreads();

  const int mbase = (lane >> 4) * 8;
  const int nloc  = lane & 15;

  for (int rep = 0; rep < MLP1_ROWREP; ++rep) {
    const int r0 = (blockIdx.x * MLP1_ROWREP + rep) * 128 + wave * 16;

    v8f acc[8];
#pragma unroll
    for (int i = 0; i < 8; ++i) acc[i] = zero8();

    v16bf a = load_a_frag(hidbf, HID_DIM, r0, 0, lane);
    for (int kk = 0; kk < HID_DIM; kk += 32) {
      v16bf anext = load_a_frag(hidbf, HID_DIM, r0, (kk + 32) & (HID_DIM - 1), lane);
      v16bf b[8];
#pragma unroll
      for (int ct = 0; ct < 8; ++ct) b[ct] = load_b_frag(sW, HID_DIM, ct * 16, kk, lane);
#pragma unroll
      for (int ct = 0; ct < 8; ++ct) acc[ct] = WMMA_BF16(a, b[ct], acc[ct]);
      a = anext;
    }

#pragma unroll
    for (int ct = 0; ct < 8; ++ct) {
      int col    = ct * 16 + nloc;
      float bias = vb[col];
#pragma unroll
      for (int i = 0; i < 8; ++i) {
        int row = r0 + mbase + i;
        float v = acc[ct][i] + bias;
        v = v > 0.0f ? v : 0.0f;
        dst[(size_t)row * MLP_DIM + col] = f2bf(sgn * v);
      }
    }
  }
}

// ------------------------------------------------------------- MLP L2
// out = H1 @ Wa2.T + (-G1) @ Wg2.T + bdelta  [N,256] f32
// grid N/256, block 256, dynamic LDS = 128KB (Wa2 + Wg2).
#define MLP2_ROWREP 4
__global__ void __launch_bounds__(256) k_mlp2(
    const bf16* __restrict__ h1, const bf16* __restrict__ g1,
    const bf16* __restrict__ wa2, const bf16* __restrict__ wg2,
    const float* __restrict__ bdelta, float* __restrict__ outf) {
  extern __shared__ char smem_raw[];
  bf16* sWa = (bf16*)smem_raw;                       // [256][128]
  bf16* sWg = sWa + (size_t)OUT_DIM * MLP_DIM;       // [256][128]

  const int lane = threadIdx.x & 31;
  const int wave = threadIdx.x >> 5;

  for (int i = threadIdx.x; i < (OUT_DIM * MLP_DIM) / 8; i += 256) {
    ((uint4*)sWa)[i] = ((const uint4*)wa2)[i];
    ((uint4*)sWg)[i] = ((const uint4*)wg2)[i];
  }
  __syncthreads();

  const int c0    = (wave >> 2) * 128;
  const int mbase = (lane >> 4) * 8;
  const int nloc  = lane & 15;

  for (int rep = 0; rep < MLP2_ROWREP; ++rep) {
    const int r0 = (blockIdx.x * MLP2_ROWREP + rep) * 64 + (wave & 3) * 16;

    v8f acc[8];
#pragma unroll
    for (int i = 0; i < 8; ++i) acc[i] = zero8();

    v16bf aH = load_a_frag(h1, MLP_DIM, r0, 0, lane);
    v16bf aG = load_a_frag(g1, MLP_DIM, r0, 0, lane);
    for (int kk = 0; kk < MLP_DIM; kk += 32) {
      int kn = (kk + 32) & (MLP_DIM - 1);
      v16bf aHn = load_a_frag(h1, MLP_DIM, r0, kn, lane);
      v16bf aGn = load_a_frag(g1, MLP_DIM, r0, kn, lane);
      v16bf bA[8], bG[8];
#pragma unroll
      for (int ct = 0; ct < 8; ++ct) {
        bA[ct] = load_b_frag(sWa, MLP_DIM, c0 + ct * 16, kk, lane);
        bG[ct] = load_b_frag(sWg, MLP_DIM, c0 + ct * 16, kk, lane);
      }
#pragma unroll
      for (int ct = 0; ct < 8; ++ct) {
        acc[ct] = WMMA_BF16(aH, bA[ct], acc[ct]);
        acc[ct] = WMMA_BF16(aG, bG[ct], acc[ct]);
      }
      aH = aHn;
      aG = aGn;
    }

#pragma unroll
    for (int ct = 0; ct < 8; ++ct) {
      int col    = c0 + ct * 16 + nloc;
      float bias = bdelta[col];
#pragma unroll
      for (int i = 0; i < 8; ++i) {
        int row = r0 + mbase + i;
        outf[(size_t)row * OUT_DIM + col] = acc[ct][i] + bias;
      }
    }
  }
}

// ----------------------------------------------------- tension + out bf16
__global__ void __launch_bounds__(256) k_tension(const float* __restrict__ outf,
                                                 bf16* __restrict__ outbf,
                                                 float* __restrict__ ten) {
  __shared__ float red[256];
  int row = blockIdx.x;
  int t   = threadIdx.x;
  float v = outf[(size_t)row * OUT_DIM + t];
  outbf[(size_t)row * OUT_DIM + t] = f2bf(v);
  red[t] = v * v;
  __syncthreads();
  for (int s = 128; s > 0; s >>= 1) {
    if (t < s) red[t] += red[t + s];
    __syncthreads();
  }
  if (t == 0) ten[row] = red[0] * (1.0f / OUT_DIM);
}

// -------------------------------------- softmax stats + mean tension
__global__ void __launch_bounds__(1024) k_red(const float* __restrict__ ten,
                                              float* __restrict__ scal,
                                              float* __restrict__ dout) {
  __shared__ float red[1024];
  int t = threadIdx.x;
  float mx = -3.4e38f;
  for (int i = t; i < N_CELLS; i += 1024) mx = fmaxf(mx, ten[i]);
  red[t] = mx;
  __syncthreads();
  for (int s = 512; s > 0; s >>= 1) {
    if (t < s) red[t] = fmaxf(red[t], red[t + s]);
    __syncthreads();
  }
  float tmax = red[0];
  __syncthreads();

  float se = 0.f, st = 0.f;
  for (int i = t; i < N_CELLS; i += 1024) {
    float v = ten[i];
    se += __expf(v - tmax);
    st += v;
  }
  red[t] = se;
  __syncthreads();
  for (int s = 512; s > 0; s >>= 1) {
    if (t < s) red[t] += red[t + s];
    __syncthreads();
  }
  float sumexp = red[0];
  __syncthreads();

  red[t] = st;
  __syncthreads();
  for (int s = 512; s > 0; s >>= 1) {
    if (t < s) red[t] += red[t + s];
    __syncthreads();
  }
  if (t == 0) {
    scal[0] = tmax;
    scal[1] = sumexp;
    dout[OUT_DIM] = red[0] * (1.0f / N_CELLS);  // mean_tension
  }
}

// ------------------------- softmax-weighted output sum (deterministic)
__global__ void __launch_bounds__(256) k_wout(const float* __restrict__ outf,
                                              const float* __restrict__ ten,
                                              const float* __restrict__ scal,
                                              float* __restrict__ partial) {
  int col = threadIdx.x;
  float tmax = scal[0];
  float p = 0.f;
  for (int row = blockIdx.x; row < N_CELLS; row += gridDim.x) {
    float w = __expf(ten[row] - tmax);
    p += w * outf[(size_t)row * OUT_DIM + col];
  }
  partial[(size_t)blockIdx.x * OUT_DIM + col] = p;
}

__global__ void __launch_bounds__(256) k_wout2(const float* __restrict__ partial,
                                               const float* __restrict__ scal,
                                               float* __restrict__ dout) {
  int col = threadIdx.x;
  float s = 0.f;
  for (int b = 0; b < 256; ++b) s += partial[(size_t)b * OUT_DIM + col];
  dout[col] = s / scal[1];
}

// ----------------------------------------------------------- fused GRU
// gates = out@W_ih[:, :256].T + hiddens@W_hh.T + t*W_ih[:,256] + biases.
// grid (N/256, HID/64), block 256, dynamic LDS = 288KB:
//   sWhh [3][64][512] bf16 (192KB) + sWih [3][64][256] bf16 (96KB).
// Each block: stage gate-column slices once, process 4 row-tiles of 64.
#define GRU_ROWREP 4
__global__ void __launch_bounds__(256) k_gru(
    const bf16* __restrict__ hidbf, const bf16* __restrict__ outbf,
    const bf16* __restrict__ whhb, const bf16* __restrict__ wihb,
    const float* __restrict__ Wih_f, const float* __restrict__ bih,
    const float* __restrict__ bhh, const float* __restrict__ hiddens,
    const float* __restrict__ ten, float* __restrict__ newh) {
  extern __shared__ char smem_raw[];
  bf16* sWhh = (bf16*)smem_raw;                         // 3 * 64 * 512
  bf16* sWih = sWhh + (size_t)3 * 64 * HID_DIM;         // 3 * 64 * 256

  const int lane  = threadIdx.x & 31;
  const int wave  = threadIdx.x >> 5;
  const int cbase = blockIdx.y * 64;

  // cooperative LDS fill of the 3 gate slices (rows cbase..cbase+63 per gate)
  for (int g = 0; g < 3; ++g) {
    const uint4* srcH = (const uint4*)(whhb + (size_t)(g * HID_DIM + cbase) * HID_DIM);
    uint4* dstH = (uint4*)(sWhh + (size_t)g * 64 * HID_DIM);
    for (int i = threadIdx.x; i < (64 * HID_DIM) / 8; i += 256) dstH[i] = srcH[i];
    const uint4* srcI = (const uint4*)(wihb + (size_t)(g * HID_DIM + cbase) * OUT_DIM);
    uint4* dstI = (uint4*)(sWih + (size_t)g * 64 * OUT_DIM);
    for (int i = threadIdx.x; i < (64 * OUT_DIM) / 8; i += 256) dstI[i] = srcI[i];
  }
  __syncthreads();

  const int cw    = (wave >> 2) * 32;  // wave's col offset within the 64-col slice
  const int mbase = (lane >> 4) * 8;
  const int nloc  = lane & 15;

  for (int rep = 0; rep < GRU_ROWREP; ++rep) {
    const int r0 = (blockIdx.x * GRU_ROWREP + rep) * 64 + (wave & 3) * 16;

    v8f accR[2], accZ[2], accIN[2], accHN[2];
#pragma unroll
    for (int i = 0; i < 2; ++i) {
      accR[i] = zero8(); accZ[i] = zero8(); accIN[i] = zero8(); accHN[i] = zero8();
    }

    // gh: hiddens @ W_hh.T  (K = 512) -> r, z, h_n
    {
      v16bf a = load_a_frag(hidbf, HID_DIM, r0, 0, lane);
      for (int kk = 0; kk < HID_DIM; kk += 32) {
        v16bf anext = load_a_frag(hidbf, HID_DIM, r0, (kk + 32) & (HID_DIM - 1), lane);
        v16bf bR[2], bZ[2], bN[2];
#pragma unroll
        for (int ct = 0; ct < 2; ++ct) {
          int cl = cw + ct * 16;
          bR[ct] = load_b_frag(sWhh,                          HID_DIM, cl, kk, lane);
          bZ[ct] = load_b_frag(sWhh + (size_t)1 * 64 * HID_DIM, HID_DIM, cl, kk, lane);
          bN[ct] = load_b_frag(sWhh + (size_t)2 * 64 * HID_DIM, HID_DIM, cl, kk, lane);
        }
#pragma unroll
        for (int ct = 0; ct < 2; ++ct) {
          accR[ct]  = WMMA_BF16(a, bR[ct], accR[ct]);
          accZ[ct]  = WMMA_BF16(a, bZ[ct], accZ[ct]);
          accHN[ct] = WMMA_BF16(a, bN[ct], accHN[ct]);
        }
        a = anext;
      }
    }
    // gi: out @ W_ih[:, :256].T  (K = 256) -> r, z, i_n
    {
      v16bf a = load_a_frag(outbf, OUT_DIM, r0, 0, lane);
      for (int kk = 0; kk < OUT_DIM; kk += 32) {
        v16bf anext = load_a_frag(outbf, OUT_DIM, r0, (kk + 32) & (OUT_DIM - 1), lane);
        v16bf bR[2], bZ[2], bN[2];
#pragma unroll
        for (int ct = 0; ct < 2; ++ct) {
          int cl = cw + ct * 16;
          bR[ct] = load_b_frag(sWih,                          OUT_DIM, cl, kk, lane);
          bZ[ct] = load_b_frag(sWih + (size_t)1 * 64 * OUT_DIM, OUT_DIM, cl, kk, lane);
          bN[ct] = load_b_frag(sWih + (size_t)2 * 64 * OUT_DIM, OUT_DIM, cl, kk, lane);
        }
#pragma unroll
        for (int ct = 0; ct < 2; ++ct) {
          accR[ct]  = WMMA_BF16(a, bR[ct], accR[ct]);
          accZ[ct]  = WMMA_BF16(a, bZ[ct], accZ[ct]);
          accIN[ct] = WMMA_BF16(a, bN[ct], accIN[ct]);
        }
        a = anext;
      }
    }

#pragma unroll
    for (int ct = 0; ct < 2; ++ct) {
      int col = cbase + cw + ct * 16 + nloc;
      float wtR = Wih_f[(size_t)col * 257 + 256];
      float wtZ = Wih_f[(size_t)(512 + col) * 257 + 256];
      float wtN = Wih_f[(size_t)(1024 + col) * 257 + 256];
      float bR  = bih[col] + bhh[col];
      float bZ  = bih[512 + col] + bhh[512 + col];
      float biN = bih[1024 + col];
      float bhN = bhh[1024 + col];
#pragma unroll
      for (int i = 0; i < 8; ++i) {
        int row = r0 + mbase + i;
        float t  = ten[row];
        float h  = hiddens[(size_t)row * HID_DIM + col];
        float rg = sigmoidf_(accR[ct][i] + t * wtR + bR);
        float zg = sigmoidf_(accZ[ct][i] + t * wtZ + bZ);
        float hn = accHN[ct][i] + bhN;
        float in = accIN[ct][i] + t * wtN + biN;
        float n  = tanhf(in + rg * hn);
        newh[(size_t)row * HID_DIM + col] = (1.0f - zg) * n + zg * h;
      }
    }
  }
}

// -------------------------------------------------------- faction mean
__global__ void __launch_bounds__(128) k_fmean(const float* __restrict__ newh,
                                               float* __restrict__ fmean) {
  int f   = blockIdx.x >> 2;
  int col = (blockIdx.x & 3) * 128 + threadIdx.x;
  const float* base = newh + (size_t)f * FS * HID_DIM;
  float s = 0.f;
  for (int r = 0; r < FS; ++r) s += base[(size_t)r * HID_DIM + col];
  fmean[f * HID_DIM + col] = s * (1.0f / FS);
}

// ---------------------------------------------------------- final blend
__global__ void k_sync(const float* __restrict__ newh,
                       const float* __restrict__ fmean, float* __restrict__ dout) {
  size_t total  = (size_t)N_CELLS * HID_DIM;
  size_t stride = (size_t)gridDim.x * blockDim.x;
  for (size_t i = (size_t)blockIdx.x * blockDim.x + threadIdx.x; i < total; i += stride) {
    int col = (int)(i & (HID_DIM - 1));
    int row = (int)(i >> 9);
    int f   = row >> 12;
    dout[OUT_DIM + 1 + i] = 0.85f * newh[i] + 0.15f * fmean[f * HID_DIM + col];
  }
}

// --------------------------------------------------------------- launch
static inline char* wsalloc(char* base, size_t& off, size_t bytes) {
  char* p = base + off;
  off += (bytes + 255) & ~(size_t)255;
  return p;
}

extern "C" void kernel_launch(void* const* d_in, const int* in_sizes, int n_in,
                              void* d_out, int out_size, void* d_ws, size_t ws_size,
                              hipStream_t stream) {
  (void)in_sizes; (void)n_in; (void)out_size; (void)ws_size;
  const float* x    = (const float*)d_in[0];
  const float* hid  = (const float*)d_in[1];
  const float* Wa1  = (const float*)d_in[2];
  const float* ba1  = (const float*)d_in[3];
  const float* Wa2  = (const float*)d_in[4];
  const float* ba2  = (const float*)d_in[5];
  const float* Wg1  = (const float*)d_in[6];
  const float* bg1  = (const float*)d_in[7];
  const float* Wg2  = (const float*)d_in[8];
  const float* bg2  = (const float*)d_in[9];
  const float* Wih  = (const float*)d_in[10];
  const float* Whh  = (const float*)d_in[11];
  const float* bih  = (const float*)d_in[12];
  const float* bhh  = (const float*)d_in[13];
  float* dout = (float*)d_out;

  char* base = (char*)d_ws;
  size_t off = 0;
  bf16*  hidbf  = (bf16*)wsalloc(base, off, (size_t)N_CELLS * HID_DIM * 2);
  bf16*  h1     = (bf16*)wsalloc(base, off, (size_t)N_CELLS * MLP_DIM * 2);
  bf16*  g1     = (bf16*)wsalloc(base, off, (size_t)N_CELLS * MLP_DIM * 2);
  float* outf   = (float*)wsalloc(base, off, (size_t)N_CELLS * OUT_DIM * 4);
  bf16*  outbf  = (bf16*)wsalloc(base, off, (size_t)N_CELLS * OUT_DIM * 2);
  float* ten    = (float*)wsalloc(base, off, (size_t)N_CELLS * 4);
  float* newh   = (float*)wsalloc(base, off, (size_t)N_CELLS * HID_DIM * 4);
  float* fmean  = (float*)wsalloc(base, off, (size_t)N_F * HID_DIM * 4);
  bf16*  wa1h   = (bf16*)wsalloc(base, off, (size_t)MLP_DIM * HID_DIM * 2);
  bf16*  wg1h   = (bf16*)wsalloc(base, off, (size_t)MLP_DIM * HID_DIM * 2);
  bf16*  wa2b   = (bf16*)wsalloc(base, off, (size_t)OUT_DIM * MLP_DIM * 2);
  bf16*  wg2b   = (bf16*)wsalloc(base, off, (size_t)OUT_DIM * MLP_DIM * 2);
  bf16*  wihb   = (bf16*)wsalloc(base, off, (size_t)3 * HID_DIM * OUT_DIM * 2);
  bf16*  whhb   = (bf16*)wsalloc(base, off, (size_t)3 * HID_DIM * HID_DIM * 2);
  float* va     = (float*)wsalloc(base, off, MLP_DIM * 4);
  float* vg     = (float*)wsalloc(base, off, MLP_DIM * 4);
  float* bdelta = (float*)wsalloc(base, off, OUT_DIM * 4);
  float* scal   = (float*)wsalloc(base, off, 64);
  float* part   = (float*)wsalloc(base, off, (size_t)256 * OUT_DIM * 4);

  // bf16 conversions
  k_f2bf<<<4096, 256, 0, stream>>>(hid, hidbf, N_CELLS * HID_DIM);
  k_f2bf_sub<<<128, 256, 0, stream>>>(Wa1, wa1h, MLP_DIM, HID_DIM, IN_DIM + HID_DIM, IN_DIM);
  k_f2bf_sub<<<128, 256, 0, stream>>>(Wg1, wg1h, MLP_DIM, HID_DIM, IN_DIM + HID_DIM, IN_DIM);
  k_f2bf_sub<<<64, 256, 0, stream>>>(Wa2, wa2b, OUT_DIM, MLP_DIM, MLP_DIM, 0);
  k_f2bf_sub<<<64, 256, 0, stream>>>(Wg2, wg2b, OUT_DIM, MLP_DIM, MLP_DIM, 0);
  k_f2bf_sub<<<512, 256, 0, stream>>>(Wih, wihb, 3 * HID_DIM, OUT_DIM, OUT_DIM + 1, 0);
  k_f2bf_sub<<<1024, 256, 0, stream>>>(Whh, whhb, 3 * HID_DIM, HID_DIM, HID_DIM, 0);

  // broadcast-x precompute
  k_xvec<<<1, 256, 0, stream>>>(x, Wa1, ba1, Wg1, bg1, ba2, bg2, va, vg, bdelta);

  // MLP (LDS-staged weights, double-buffered A)
  size_t lds_mlp1 = (size_t)MLP_DIM * HID_DIM * 2;          // 128KB
  size_t lds_mlp2 = (size_t)2 * OUT_DIM * MLP_DIM * 2;      // 128KB
  k_mlp1<<<dim3(N_CELLS / (128 * MLP1_ROWREP), 2), 256, lds_mlp1, stream>>>(
      hidbf, wa1h, wg1h, va, vg, h1, g1);
  k_mlp2<<<N_CELLS / (64 * MLP2_ROWREP), 256, lds_mlp2, stream>>>(
      h1, g1, wa2b, wg2b, bdelta, outf);

  // tension + bf16 out
  k_tension<<<N_CELLS, 256, 0, stream>>>(outf, outbf, ten);

  // softmax stats / mean tension, weighted output (deterministic two-pass)
  k_red<<<1, 1024, 0, stream>>>(ten, scal, dout);
  k_wout<<<256, 256, 0, stream>>>(outf, ten, scal, part);
  k_wout2<<<1, 256, 0, stream>>>(part, scal, dout);

  // fused GRU (LDS-staged gate weight slices)
  size_t lds_gru = (size_t)3 * 64 * HID_DIM * 2 + (size_t)3 * 64 * OUT_DIM * 2;  // 288KB
  k_gru<<<dim3(N_CELLS / (64 * GRU_ROWREP), HID_DIM / 64), 256, lds_gru, stream>>>(
      hidbf, outbf, whhb, wihb, Wih, bih, bhh, hid, ten, newh);

  // faction smoothing into d_out
  k_fmean<<<N_F * 4, 128, 0, stream>>>(newh, fmean);
  k_sync<<<8192, 256, 0, stream>>>(newh, fmean, dout);
}
